// train_75488345194687
// MI455X (gfx1250) — matrix operand order repaired and checked by reading, not compiled, not run on hardware
//
#include <hip/hip_runtime.h>

typedef __attribute__((ext_vector_type(16))) _Float16 v16h;
typedef __attribute__((ext_vector_type(8)))  float    v8f;

constexpr int   kPeriod = 30;
constexpr float kKappa = 0.2f, kYbar = 2.0f, kB = 1.0f, kC = 0.1f;
constexpr float kGamma = 1.2f, kRho = 0.03f, kSigma = 0.03f, kDT = 0.005f;

__device__ __forceinline__ v8f wmma16(v16h a, v16h b, v8f c) {
  // v_wmma_f32_16x16x32_f16 : D = A(16x32 f16) * B(32x16 f16) + C(f32)
  return __builtin_amdgcn_wmma_f32_16x16x32_f16(false, a, false, b, (short)0, c,
                                                false, false);
}

__device__ __forceinline__ float fast_tanh(float x) {
#if __has_builtin(__builtin_amdgcn_tanhf)
  return __builtin_amdgcn_tanhf(x);      // v_tanh_f32 (TRANS, 1 instr)
#else
  x = fminf(fmaxf(x, -15.0f), 15.0f);
  float e = __expf(-2.0f * x);
  return (1.0f - e) * __builtin_amdgcn_rcpf(1.0f + e);
#endif
}

__device__ __forceinline__ unsigned hash_u32(unsigned x) {
  x ^= x >> 17; x *= 0xed5ad4bbu;
  x ^= x >> 11; x *= 0xac4c1b51u;
  x ^= x >> 15; x *= 0x31848babu;
  x ^= x >> 14;
  return x;
}

union TileCast { v16h v; uint2 u[4]; };

// A-tile LDS image: 12 tiles, per tile 4 uint2-chunks x 32 lanes, lane-major
// (byte addr = lane*8 within a chunk -> banks 2L,2L+1: conflict-free b64).
#define AT_IDX(h, m, kt, p) ((((h) * 4 + (m) * 2 + (kt)) * 4 + (p)) * 32)

// Evaluate q1..q3 and sigma-scaled JVP sensitivities for the 32 samples owned
// by this wave.  Lane L owns sample (waveBase + L).
__device__ __forceinline__ void eval_net(
    float Y2, float Y3, int lane,
    const uint2* __restrict__ s_at,           // Wh^T tiles (A layout) in LDS
    const float* __restrict__ w0r0, const float* __restrict__ w0r1,
    const float* __restrict__ b0,
    const float2* __restrict__ bhwo,          // [3][32] = (bh, wo)
    const float* __restrict__ bo,
    float (&q)[3], float (&sA)[3], float (&sB)[3]) {
  // Broadcast Y of sample groups nt=0 (lanes 0..15) / nt=1 (lanes 16..31)
  float y2s[2], y3s[2];
  y2s[0] = __shfl(Y2, lane & 15, 32);
  y2s[1] = __shfl(Y2, 16 | (lane & 15), 32);
  y3s[0] = __shfl(Y3, lane & 15, 32);
  y3s[1] = __shfl(Y3, 16 | (lane & 15), 32);

  // First layer (2->64) on VALU; pack B-tiles (K=hid, N=sample) in the 16-bit
  // B layout: lane L: N = nt*16 + (L&15), K = kt*32 + ((i&8)?16:0) + ((L&16)?8:0) + (i&7).
  v16h Bf[2][2], Bd2[2][2], Bd3[2][2];
  const int kb = lane & 16;
#pragma unroll
  for (int kt = 0; kt < 2; ++kt) {
#pragma unroll
    for (int i = 0; i < 16; ++i) {
      const int k = kt * 32 + ((i & 8) ? 16 : 0) + (kb >> 1) + (i & 7);
      const float w00 = w0r0[k], w01 = w0r1[k], bb = b0[k];
#pragma unroll
      for (int nt = 0; nt < 2; ++nt) {
        float a  = fmaf(y2s[nt], w00, fmaf(y3s[nt], w01, bb));
        float hv = fast_tanh(a);
        float g  = 1.0f - hv * hv;
        Bf[kt][nt][i]  = (_Float16)hv;          // h
        Bd2[kt][nt][i] = (_Float16)(g * w00);   // dh/dY2
        Bd3[kt][nt][i] = (_Float16)(g * w01);   // dh/dY3
      }
    }
  }

  const int rb = (lane & 16) ? 8 : 0;           // C/D row base for this lane half
#pragma unroll
  for (int h = 0; h < 3; ++h) {
    // keep this head's A-tile LDS loads from being hoisted/coalesced upward
    asm volatile("" ::: "memory");
    v16h Ah[2][2];                              // 32 VGPRs staging, per head
#pragma unroll
    for (int m = 0; m < 2; ++m)
#pragma unroll
      for (int kt = 0; kt < 2; ++kt) {
        TileCast tc;
#pragma unroll
        for (int p = 0; p < 4; ++p)
          tc.u[p] = s_at[AT_IDX(h, m, kt, p) + lane];
        Ah[m][kt] = tc.v;
      }

    // ---- forward pass: a1 = h @ Wh (computed as Wh^T x h^T) ----
    v8f accf[2][2];
#pragma unroll
    for (int m = 0; m < 2; ++m)
#pragma unroll
      for (int nt = 0; nt < 2; ++nt) {
        v8f c = {};
        c = wmma16(Ah[m][0], Bf[0][nt], c);
        c = wmma16(Ah[m][1], Bf[1][nt], c);
        accf[m][nt] = c;
      }
    float pu0 = 0.0f, pu1 = 0.0f;
    v8f gw[2][2];                               // (1-t^2)*wo per unit
#pragma unroll
    for (int m = 0; m < 2; ++m)
#pragma unroll
      for (int r = 0; r < 8; ++r) {
        const float2 bw = bhwo[h * 32 + m * 16 + rb + r];
        float t0 = fast_tanh(accf[m][0][r] + bw.x);
        float t1 = fast_tanh(accf[m][1][r] + bw.x);
        pu0 = fmaf(t0, bw.y, pu0);
        pu1 = fmaf(t1, bw.y, pu1);
        gw[m][0][r] = (1.0f - t0 * t0) * bw.y;
        gw[m][1][r] = (1.0f - t1 * t1) * bw.y;
      }
    float u0 = pu0 + __shfl_xor(pu0, 16, 32);   // complete 32-unit dot product
    float u1 = pu1 + __shfl_xor(pu1, 16, 32);
    q[h] = __expf(((lane < 16) ? u0 : u1) + bo[h]);

    // ---- JVP along Y2 ----
    {
      float p0 = 0.0f, p1 = 0.0f;
#pragma unroll
      for (int m = 0; m < 2; ++m) {
        v8f c0 = {}, c1 = {};
        c0 = wmma16(Ah[m][0], Bd2[0][0], c0);
        c0 = wmma16(Ah[m][1], Bd2[1][0], c0);
        c1 = wmma16(Ah[m][0], Bd2[0][1], c1);
        c1 = wmma16(Ah[m][1], Bd2[1][1], c1);
#pragma unroll
        for (int r = 0; r < 8; ++r) {
          p0 = fmaf(gw[m][0][r], c0[r], p0);
          p1 = fmaf(gw[m][1][r], c1[r], p1);
        }
      }
      float d0 = p0 + __shfl_xor(p0, 16, 32);
      float d1 = p1 + __shfl_xor(p1, 16, 32);
      sA[h] = kSigma * ((lane < 16) ? d0 : d1); // s = dq*SIGMA/q = SIGMA*da2
    }
    // ---- JVP along Y3 ----
    {
      float p0 = 0.0f, p1 = 0.0f;
#pragma unroll
      for (int m = 0; m < 2; ++m) {
        v8f c0 = {}, c1 = {};
        c0 = wmma16(Ah[m][0], Bd3[0][0], c0);
        c0 = wmma16(Ah[m][1], Bd3[1][0], c0);
        c1 = wmma16(Ah[m][0], Bd3[0][1], c1);
        c1 = wmma16(Ah[m][1], Bd3[1][1], c1);
#pragma unroll
        for (int r = 0; r < 8; ++r) {
          p0 = fmaf(gw[m][0][r], c0[r], p0);
          p1 = fmaf(gw[m][1][r], c1[r], p1);
        }
      }
      float d0 = p0 + __shfl_xor(p0, 16, 32);
      float d1 = p1 + __shfl_xor(p1, 16, 32);
      sB[h] = kSigma * ((lane < 16) ? d0 : d1);
    }
  }
}

__global__ __launch_bounds__(256) void deep_sde_wmma_kernel(
    const float* __restrict__ Y,
    const float* __restrict__ W0g, const float* __restrict__ b0g,
    const float* __restrict__ Wh1, const float* __restrict__ bh1,
    const float* __restrict__ Wo1, const float* __restrict__ bo1,
    const float* __restrict__ Wh2, const float* __restrict__ bh2,
    const float* __restrict__ Wo2, const float* __restrict__ bo2,
    const float* __restrict__ Wh3, const float* __restrict__ bh3,
    const float* __restrict__ Wo3, const float* __restrict__ bo3,
    float* __restrict__ out, int N, float lscale) {
  __shared__ float s_wh[3][64 * 32];          // 24 KB (build-time only)
  __shared__ uint2 s_at[12 * 4 * 32];         // 12 KB A-tile image (f16)
  __shared__ float s_w0r0[64], s_w0r1[64], s_b0[64];
  __shared__ float2 s_bhwo[3 * 32];           // (bh, wo)
  __shared__ float s_bo[3];
  __shared__ float s_red[8];

  const int tid  = threadIdx.x;
  const int lane = tid & 31;

  for (int i = tid; i < 64; i += 256) {
    s_w0r0[i] = W0g[i];
    s_w0r1[i] = W0g[64 + i];
    s_b0[i]   = b0g[i];
  }
  for (int i = tid; i < 2048; i += 256) {
    s_wh[0][i] = Wh1[i];
    s_wh[1][i] = Wh2[i];
    s_wh[2][i] = Wh3[i];
  }
  if (tid < 32) {
    s_bhwo[0 * 32 + tid] = make_float2(bh1[tid], Wo1[tid]);
    s_bhwo[1 * 32 + tid] = make_float2(bh2[tid], Wo2[tid]);
    s_bhwo[2 * 32 + tid] = make_float2(bh3[tid], Wo3[tid]);
  }
  if (tid == 0) { s_bo[0] = bo1[0]; s_bo[1] = bo2[0]; s_bo[2] = bo3[0]; }
  __syncthreads();

  // One wave gathers Wh^T into the 16-bit WMMA A layout and publishes it to
  // LDS (contents depend only on lane, so all waves share one image):
  // lane L: row M = m*16 + (L&15), K = kt*32 + ((i&8)?16:0) + ((L&16)?8:0) + (i&7)
  if (tid < 32) {
    const int hi8 = (lane & 16) ? 8 : 0;
#pragma unroll
    for (int h = 0; h < 3; ++h)
#pragma unroll
      for (int m = 0; m < 2; ++m)
#pragma unroll
        for (int kt = 0; kt < 2; ++kt) {
          TileCast tc;
#pragma unroll
          for (int i = 0; i < 16; ++i) {
            const int k = kt * 32 + ((i & 8) ? 16 : 0) + hi8 + (i & 7);
            const int u = m * 16 + (lane & 15);
            tc.v[i] = (_Float16)s_wh[h][k * 32 + u];
          }
#pragma unroll
          for (int p = 0; p < 4; ++p)
            s_at[AT_IDX(h, m, kt, p) + lane] = tc.u[p];
        }
  }
  __syncthreads();

  const int s = blockIdx.x * 256 + tid;       // lane L of a wave owns sample base+L
  const bool valid = (s < N);
  float Y2 = valid ? Y[2 * (size_t)s + 0] : 1.0f;
  float Y3 = valid ? Y[2 * (size_t)s + 1] : 1.0f;

  float q[3], sa[3], sb[3];
  eval_net(Y2, Y3, lane, s_at, s_w0r0, s_w0r1, s_b0, s_bhwo, s_bo, q, sa, sb);
  float q1 = q[0], q2 = q[1], q3 = q[2];
  float s11 = sa[0], s21 = sa[1], s31 = sa[2];
  float s12 = sb[0], s22 = sb[1], s32 = sb[2];
  float loss = 0.0f;

  const float sqrtDT = 0.07071067811865475f;  // sqrt(0.005)
  const float gfac   = kGamma / (kGamma - 1.0f);

#pragma unroll 1
  for (int t = 0; t < kPeriod; ++t) {
    // deterministic pseudo-noise (Box-Muller)
    unsigned base = (unsigned)s * 2654435761u ^ (unsigned)(t + 1) * 0x9E3779B9u;
    unsigned r1 = hash_u32(base);
    unsigned r2 = hash_u32(base ^ 0x68bc21ebu);
    float u1 = fmaxf((float)(r1 >> 8) * 5.9604645e-08f, 1e-7f);
    float u2 = (float)(r2 >> 8) * 5.9604645e-08f;
    float rad = sqrtf(-2.0f * __logf(u1));
    float dZ1 = sqrtDT * rad * __cosf(6.2831853f * u2);
    float dZ2 = sqrtDT * rad * __sinf(6.2831853f * u2);

    // ---- elementwise SDE step (per-lane scalars) ----
    float W = q1 + q2 + q3;
    float invW = 1.0f / W;
    float q1h = q1 * invW, q2h = q2 * invW;
    float q3h = 1.0f - q1h - q2h;
    float q1sq = q1 * q1;
    float cden = kB * q1 - kC * q1sq + Y2 + Y3;
    float invc = 1.0f / cden;
    float bq   = kB * q1 - 2.0f * kC * q1sq;
    float sc1  = fmaf(bq, s11, kSigma);
    float sc2  = fmaf(bq, s12, kSigma);
    float sW1  = q1 * s11 + q2 * s21 + q3 * s31;
    float sW2  = q1 * s12 + q2 * s22 + q3 * s32;
    float sJ1  = gfac * (sc1 * invc - sW1 * invW);
    float sJ2  = gfac * (sc2 * invc - sW2 * invW);
    float d11 = s11 * s11 + s12 * s12;
    float d12 = s11 * s21 + s12 * s22;
    float d13 = s11 * s31 + s12 * s32;
    float d22 = s21 * s21 + s22 * s22;
    float d23 = s21 * s31 + s22 * s32;
    float d33 = s31 * s31 + s32 * s32;
    float omg = 1.0f - kGamma;
    float p1 = kGamma * (q1h * d11 + q2h * d12 + q3h * d13) - omg * (sJ1 * s11 + sJ2 * s12);
    float p2 = kGamma * (q1h * d12 + q2h * d22 + q3h * d23) - omg * (sJ1 * s21 + sJ2 * s22);
    float p3 = kGamma * (q1h * d13 + q2h * d23 + q3h * d33) - omg * (sJ1 * s31 + sJ2 * s32);
    float muY2 = kKappa * (kYbar - Y2);
    float muY3 = kKappa * (kYbar - Y3);
    float r = kRho +
              kGamma * (bq * (p1 - (kB - kC * q1)) - kC * q1sq * d11 + muY2 + muY3) * invc -
              kGamma * (kGamma + 1.0f) * 0.5f * (sc1 * sc1 + sc2 * sc2) * invc * invc;
    r = r / (1.0f - kGamma * bq * invc);
    float mu1 = r + p1 - (kB - kC * q1);
    float mu2 = r + p2 - Y2 / q2;
    float mu3 = r + p3 - Y3 / q3;
    float q1n = q1 * (1.0f + mu1 * kDT + s11 * dZ1 + s12 * dZ2);
    float q2n = q2 * (1.0f + mu2 * kDT + s21 * dZ1 + s22 * dZ2);
    float q3n = q3 * (1.0f + mu3 * kDT + s31 * dZ1 + s32 * dZ2);
    float Y2n = Y2 + muY2 * kDT + kSigma * dZ1;
    float Y3n = Y3 + muY3 * kDT + kSigma * dZ2;

    // ---- re-evaluate network at the new state (WMMA-heavy) ----
    eval_net(Y2n, Y3n, lane, s_at, s_w0r0, s_w0r1, s_b0, s_bhwo, s_bo,
             q, sa, sb);
    float e1 = q1n - q[0], e2 = q2n - q[1], e3 = q3n - q[2];
    float contrib = (e1 * e1 + e2 * e2 + e3 * e3) * lscale;
    loss += valid ? contrib : 0.0f;

    q1 = q1n; q2 = q2n; q3 = q3n; Y2 = Y2n; Y3 = Y3n;
    s11 = sa[0]; s21 = sa[1]; s31 = sa[2];
    s12 = sb[0]; s22 = sb[1]; s32 = sb[2];
  }

  // wave -> block -> global reduction of the scalar loss
  loss += __shfl_xor(loss, 1, 32);
  loss += __shfl_xor(loss, 2, 32);
  loss += __shfl_xor(loss, 4, 32);
  loss += __shfl_xor(loss, 8, 32);
  loss += __shfl_xor(loss, 16, 32);
  const int wid = tid >> 5;
  if (lane == 0) s_red[wid] = loss;
  __syncthreads();
  if (tid == 0) {
    float tot = 0.0f;
#pragma unroll
    for (int i = 0; i < 8; ++i) tot += s_red[i];
    atomicAdd(out, tot);
  }
}

extern "C" void kernel_launch(void* const* d_in, const int* in_sizes, int n_in,
                              void* d_out, int out_size, void* d_ws, size_t ws_size,
                              hipStream_t stream) {
  (void)n_in; (void)out_size; (void)d_ws; (void)ws_size;
  const float* Y   = (const float*)d_in[0];
  const float* W0  = (const float*)d_in[1];
  const float* b0  = (const float*)d_in[2];
  const float* Wh1 = (const float*)d_in[3];
  const float* bh1 = (const float*)d_in[4];
  const float* Wo1 = (const float*)d_in[5];
  const float* bo1 = (const float*)d_in[6];
  const float* Wh2 = (const float*)d_in[7];
  const float* bh2 = (const float*)d_in[8];
  const float* Wo2 = (const float*)d_in[9];
  const float* bo2 = (const float*)d_in[10];
  const float* Wh3 = (const float*)d_in[11];
  const float* bh3 = (const float*)d_in[12];
  const float* Wo3 = (const float*)d_in[13];
  const float* bo3 = (const float*)d_in[14];

  const int N = in_sizes[0] / 2;
  const float lscale = 1.0f / ((float)N * (float)kPeriod);

  hipMemsetAsync(d_out, 0, sizeof(float), stream);
  const int blocks = (N + 255) / 256;
  deep_sde_wmma_kernel<<<blocks, 256, 0, stream>>>(
      Y, W0, b0, Wh1, bh1, Wo1, bo1, Wh2, bh2, Wo2, bo2, Wh3, bh3, Wo3, bo3,
      (float*)d_out, N, lscale);
}
